// FeatureEncoder_72524817760392
// MI455X (gfx1250) — compile-verified, matrix operand order checked by reading, not run on hardware
//
#include <hip/hip_runtime.h>
#include <hip/hip_bf16.h>
#include <math.h>

// ---------------------------------------------------------------------------
// CDNA5 (gfx1250) pix2pixHD-style backbone + segment mean.
// Convs/deconvs run as implicit GEMM on wave32 WMMA (v_wmma_f32_16x16x32_f16,
// f32 accumulate).  Activations flow in f16 between layers; weights are
// pre-packed into the exact WMMA B-matrix VGPR layout.  Each wave gathers one
// 16x32 A fragment and reuses it for NBLK WMMAs (16 x 16*NBLK output tile).
// ---------------------------------------------------------------------------

typedef __attribute__((ext_vector_type(16))) _Float16 v16h;
typedef __attribute__((ext_vector_type(8)))  float    v8f;

#define NUM_INST 32

// -------------------------- implicit-GEMM A fetch ---------------------------
// MODE: 0 = zero-pad conv, 1 = reflect-pad conv (stride 1),
//       2 = deconv (ConvTranspose2d k=3 s=2 p=1 output_padding=1)
template <int MODE, int KK, int STRIDE, int PAD>
__device__ __forceinline__ _Float16 fetch16(const _Float16* __restrict__ in,
                                            bool mvalid, int bbase,
                                            int y, int x, int k, int K_total,
                                            int Hin, int Win)
{
    constexpr int KHW = KK * KK;
    if (!mvalid || k >= K_total) return (_Float16)0.0f;
    int ci = k / KHW;            // compile-time divisor -> magic multiply
    int r  = k - ci * KHW;
    int ky = r / KK;
    int kx = r - ky * KK;
    int iy, ix;
    if (MODE == 2) {
        // out[y] gets in[iy] where y = 2*iy - 1 + ky  ->  iy = (y+1-ky)/2
        int ty = y + 1 - ky;
        int tx = x + 1 - kx;
        if (ty < 0 || tx < 0 || (ty & 1) || (tx & 1)) return (_Float16)0.0f;
        iy = ty >> 1; ix = tx >> 1;
        if (iy >= Hin || ix >= Win) return (_Float16)0.0f;
    } else {
        iy = y * STRIDE - PAD + ky;
        ix = x * STRIDE - PAD + kx;
        if (MODE == 1) {         // reflect padding (pad=3 < H)
            iy = iy < 0 ? -iy : (iy >= Hin ? 2 * Hin - 2 - iy : iy);
            ix = ix < 0 ? -ix : (ix >= Win ? 2 * Win - 2 - ix : ix);
        } else {
            if (iy < 0 || iy >= Hin || ix < 0 || ix >= Win) return (_Float16)0.0f;
        }
    }
    return in[(size_t)bbase + ((size_t)ci * Hin + iy) * Win + ix];
}

// ----------------------------- WMMA conv kernel -----------------------------
template <int MODE, int KK, int STRIDE, int PAD, int NBLK>
__global__ void conv_wmma_t(const unsigned short* __restrict__ in_u16,
                            const unsigned short* __restrict__ wpk_u16,
                            const float* __restrict__ bias,
                            float* __restrict__ out,
                            int B, int Cin, int Hin, int Win,
                            int Cout, int Hout, int Wout,
                            int mtiles, int ntgroups, int nkt)
{
    const int wave = blockIdx.x * (blockDim.x >> 5) + (threadIdx.x >> 5);
    const int lane = threadIdx.x & 31;
    if (wave >= mtiles * ntgroups) return;     // whole-wave exit: EXEC stays full
    const int ng  = wave % ntgroups;
    const int mt  = wave / ntgroups;
    const int nt0 = ng * NBLK;

    const _Float16* in = (const _Float16*)in_u16;
    const int M_total = B * Hout * Wout;
    const int K_total = Cin * KK * KK;
    const int HW = Hout * Wout;

    // hoisted per-lane M decode (loop-invariant)
    const int  m      = mt * 16 + (lane & 15);
    const bool mvalid = m < M_total;
    const int  mc     = mvalid ? m : 0;
    const int  b      = mc / HW;
    const int  rem    = mc - b * HW;
    const int  y      = rem / Wout;
    const int  x      = rem - y * Wout;
    const int  bbase  = b * Cin * Hin * Win;
    const int  half8  = (lane >> 4) << 3;      // 0 or 8: lane-half K offset

    const _Float16* wl[NBLK];
#pragma unroll
    for (int t = 0; t < NBLK; ++t)
        wl[t] = (const _Float16*)wpk_u16 +
                ((size_t)((nt0 + t) * nkt) * 32 + lane) * 16;

    v8f acc[NBLK];
#pragma unroll
    for (int t = 0; t < NBLK; ++t) { v8f z = {}; acc[t] = z; }

    for (int kt = 0; kt < nkt; ++kt) {
        v16h bv[NBLK];
#pragma unroll
        for (int t = 0; t < NBLK; ++t) {
            bv[t] = *(const v16h*)wl[t];
            wl[t] += 32 * 16;                  // next 32x16 K-tile
            __builtin_prefetch(wl[t], 0, 1);   // global_prefetch_b8 (speculative)
        }
        const int kbase = kt * 32 + half8;
        v16h a;
#pragma unroll
        for (int j = 0; j < 8; ++j) {
            // A-matrix 16x32 f16 layout: VGPR j<4 -> K = half8 + 2j{,+1};
            //                            VGPR j>=4 -> K = 16 + half8 + 2(j-4){,+1}
            int k = kbase + ((j < 4) ? 2 * j : 16 + 2 * (j - 4));
            a[2 * j]     = fetch16<MODE, KK, STRIDE, PAD>(in, mvalid, bbase, y, x,
                                                          k,     K_total, Hin, Win);
            a[2 * j + 1] = fetch16<MODE, KK, STRIDE, PAD>(in, mvalid, bbase, y, x,
                                                          k + 1, K_total, Hin, Win);
        }
#pragma unroll
        for (int t = 0; t < NBLK; ++t)
            acc[t] = __builtin_amdgcn_wmma_f32_16x16x32_f16(
                /*neg_a=*/false, a, /*neg_b=*/false, bv[t],
                /*c_mod=*/(short)0, acc[t], /*reuse_a=*/false, /*reuse_b=*/false);
    }

    // Epilogue: C layout -> lane n = lane&15; VGPR r holds M = r + 8*(lane>>4)
    const int ncol = lane & 15;
#pragma unroll
    for (int t = 0; t < NBLK; ++t) {
        const int n  = (nt0 + t) * 16 + ncol;
        const float bn = (n < Cout) ? bias[n] : 0.0f;
#pragma unroll
        for (int r = 0; r < 8; ++r) {
            int mm = mt * 16 + r + half8;
            if (mm < M_total && n < Cout) {
                int bb = mm / HW;
                int rr = mm - bb * HW;
                int yy = rr / Wout;
                int xx = rr - yy * Wout;
                out[(((size_t)bb * Cout + n) * Hout + yy) * Wout + xx] = acc[t][r] + bn;
            }
        }
    }
}

// ------------------------- weight pack (B layout) ---------------------------
// Packs f32 weights (OIHW for conv, (Cin,Cout,kh,kw) for deconv) into f16 in
// the WMMA B-matrix layout, zero-padded to Cout%16 / K%32.
__global__ void pack_w_kernel(const float* __restrict__ w,
                              unsigned short* __restrict__ wpk_u16,
                              int Cout, int Cin, int KHW, int deconv,
                              int ntiles, int nkt)
{
    int idx   = blockIdx.x * blockDim.x + threadIdx.x;
    int total = ntiles * nkt * 32 * 16;
    if (idx >= total) return;
    int e    = idx & 15;
    int lane = (idx >> 4) & 31;
    int tk   = idx >> 9;              // = nt*nkt + kt
    int kt   = tk % nkt;
    int nt   = tk / nkt;
    int j = e >> 1, t = e & 1;
    int k = kt * 32 + ((lane >> 4) << 3) + ((j < 4) ? 2 * j : 16 + 2 * (j - 4)) + t;
    int n = nt * 16 + (lane & 15);
    int K_total = Cin * KHW;
    float v = 0.0f;
    if (n < Cout && k < K_total) {
        int ci = k / KHW;
        int r  = k - ci * KHW;
        v = deconv ? w[((size_t)ci * Cout + n) * KHW + r]
                   : w[((size_t)n * Cin + ci) * KHW + r];
    }
    _Float16 h = (_Float16)v;
    wpk_u16[idx] = *(unsigned short*)&h;
}

// --------------------------- instance-norm stats ----------------------------
__global__ void inorm_stat_kernel(const float* __restrict__ x,
                                  float* __restrict__ stats, int HW)
{
    const int bc = blockIdx.x;
    const float* p = x + (size_t)bc * HW;
    float s = 0.0f, ss = 0.0f;
    for (int i = threadIdx.x; i < HW; i += blockDim.x) {
        float v = p[i];
        s += v; ss += v * v;
    }
    __shared__ float sh[256], sh2[256];
    sh[threadIdx.x] = s;  sh2[threadIdx.x] = ss;
    __syncthreads();
    for (int o = blockDim.x >> 1; o > 0; o >>= 1) {
        if (threadIdx.x < o) {
            sh[threadIdx.x]  += sh[threadIdx.x + o];
            sh2[threadIdx.x] += sh2[threadIdx.x + o];
        }
        __syncthreads();
    }
    if (threadIdx.x == 0) {
        float mean = sh[0] / (float)HW;
        float var  = sh2[0] / (float)HW - mean * mean;
        stats[2 * bc]     = mean;
        stats[2 * bc + 1] = rsqrtf(var + 1e-5f);
    }
}

// normalize + ReLU, emitting next-layer activations directly in f16
__global__ void inorm_relu_h16_kernel(const float* __restrict__ x,
                                      const float* __restrict__ stats,
                                      unsigned short* __restrict__ o16,
                                      int HW, long total)
{
    long i = (long)blockIdx.x * blockDim.x + threadIdx.x;
    if (i >= total) return;
    int bc = (int)(i / HW);
    float v = (x[i] - stats[2 * bc]) * stats[2 * bc + 1];
    v = v > 0.0f ? v : 0.0f;
    _Float16 h = (_Float16)v;
    o16[i] = *(unsigned short*)&h;
}

__global__ void f32_to_h16_kernel(const float* __restrict__ x,
                                  unsigned short* __restrict__ o16, long total)
{
    long i = (long)blockIdx.x * blockDim.x + threadIdx.x;
    if (i >= total) return;
    _Float16 h = (_Float16)x[i];
    o16[i] = *(unsigned short*)&h;
}

__global__ void tanh_kernel(float* __restrict__ x, long total)
{
    long i = (long)blockIdx.x * blockDim.x + threadIdx.x;
    if (i >= total) return;
    x[i] = tanhf(x[i]);
}

// ------------------------------ segment mean --------------------------------
__global__ void seg_zero_kernel(float* __restrict__ seg)
{
    int i = threadIdx.x;
    if (i < NUM_INST * 4) seg[i] = 0.0f;
}

// seg layout: [NUM_INST][4] = {c0_sum, c1_sum, c2_sum, count}
__global__ void seg_acc_kernel(const float* __restrict__ feat,
                               const int* __restrict__ inst,
                               float* __restrict__ seg, int B, int HW)
{
    __shared__ float ls[NUM_INST * 4];
    for (int i = threadIdx.x; i < NUM_INST * 4; i += blockDim.x) ls[i] = 0.0f;
    __syncthreads();
    const long total = (long)B * HW;
    for (long p = (long)blockIdx.x * blockDim.x + threadIdx.x; p < total;
         p += (long)gridDim.x * blockDim.x) {
        int b  = (int)(p / HW);
        int hw = (int)(p - (long)b * HW);
        int s  = inst[p] & (NUM_INST - 1);       // inst flat [B*H*W]
        float f0 = feat[((size_t)b * 3 + 0) * HW + hw];
        float f1 = feat[((size_t)b * 3 + 1) * HW + hw];
        float f2 = feat[((size_t)b * 3 + 2) * HW + hw];
        atomicAdd(&ls[s * 4 + 0], f0);
        atomicAdd(&ls[s * 4 + 1], f1);
        atomicAdd(&ls[s * 4 + 2], f2);
        atomicAdd(&ls[s * 4 + 3], 1.0f);
    }
    __syncthreads();
    for (int i = threadIdx.x; i < NUM_INST * 4; i += blockDim.x)
        atomicAdd(&seg[i], ls[i]);
}

__global__ void seg_out_kernel(const int* __restrict__ inst,
                               const float* __restrict__ seg,
                               float* __restrict__ out, int B, int HW)
{
    long total = (long)B * 3 * HW;
    long i = (long)blockIdx.x * blockDim.x + threadIdx.x;
    if (i >= total) return;
    int b   = (int)(i / (3L * HW));
    long r  = i - (long)b * 3 * HW;
    int c   = (int)(r / HW);
    int hw  = (int)(r - (long)c * HW);
    int s   = inst[(long)b * HW + hw] & (NUM_INST - 1);
    float cnt = seg[s * 4 + 3];
    cnt = cnt > 1.0f ? cnt : 1.0f;
    out[i] = seg[s * 4 + c] / cnt;
}

// ---------------------------------------------------------------------------
extern "C" void kernel_launch(void* const* d_in, const int* in_sizes, int n_in,
                              void* d_out, int out_size, void* d_ws, size_t ws_size,
                              hipStream_t stream)
{
    (void)in_sizes; (void)n_in; (void)out_size; (void)ws_size;
    const float* input = (const float*)d_in[0];
    const int*   inst  = (const int*)d_in[1];
    const float* W[10];
    const float* Bb[10];
    for (int i = 0; i < 10; ++i) {
        W[i]  = (const float*)d_in[2 + 2 * i];
        Bb[i] = (const float*)d_in[3 + 2 * i];
    }

    // ---- layer table
    static const int LCin[10]  = {  3, 16, 32,  64, 128, 256, 128, 64, 32, 16};
    static const int LHin[10]  = {512,512,256, 128,  64,  32,  64,128,256,512};
    static const int LCout[10] = { 16, 32, 64, 128, 256, 128,  64, 32, 16,  3};
    static const int LHout[10] = {512,256,128,  64,  32,  64, 128,256,512,512};
    static const int LK[10]    = {  7,  3,  3,   3,   3,   3,   3,  3,  3,  7};
    static const int LM[10]    = {  1,  0,  0,   0,   0,   2,   2,  2,  2,  1};
    // N-register-blocking per layer (ntiles: 1,2,4,8,16,8,4,2,1,1)
    static const int LNB[10]   = {  1,  2,  4,   4,   4,   4,   4,  2,  1,  1};
    const int Bn = 4;

    // ---- workspace layout
    char*  ws  = (char*)d_ws;
    size_t off = 0;
    const size_t f32Bytes = (size_t)Bn * 16 * 512 * 512 * sizeof(float);          // 67.1 MB
    const size_t h16Bytes = (size_t)Bn * 16 * 512 * 512 * sizeof(unsigned short); // 33.6 MB
    float*          convOut = (float*)(ws + off);          off += f32Bytes;
    unsigned short* hbuf[2];
    hbuf[0] = (unsigned short*)(ws + off);                 off += h16Bytes;
    hbuf[1] = (unsigned short*)(ws + off);                 off += h16Bytes;

    unsigned short* wpk[10];
    int ntiles[10], nkt[10];
    for (int L = 0; L < 10; ++L) {
        int Kt = LCin[L] * LK[L] * LK[L];
        ntiles[L] = (LCout[L] + 15) / 16;
        nkt[L]    = (Kt + 31) / 32;
        wpk[L]    = (unsigned short*)(ws + off);
        size_t bytes = (size_t)ntiles[L] * nkt[L] * 32 * 16 * sizeof(unsigned short);
        off += (bytes + 255) & ~(size_t)255;
    }
    float* stats = (float*)(ws + off); off += ((size_t)2 * Bn * 256 * 4 + 255) & ~(size_t)255;
    float* seg   = (float*)(ws + off); off += 256;

    // ---- network input -> f16
    {
        long tot = (long)Bn * 3 * 512 * 512;
        f32_to_h16_kernel<<<(int)((tot + 255) / 256), 256, 0, stream>>>(
            input, hbuf[0], tot);
    }

    // ---- backbone
    int cur = 0;
    for (int L = 0; L < 10; ++L) {
        const int Cin = LCin[L], Hin = LHin[L], Win = LHin[L];
        const int Cout = LCout[L], Hout = LHout[L], Wout = LHout[L];
        const int mode = LM[L], nblk = LNB[L];

        {   // pack weights into WMMA B layout (tiny, deterministic)
            int total = ntiles[L] * nkt[L] * 32 * 16;
            pack_w_kernel<<<(total + 255) / 256, 256, 0, stream>>>(
                W[L], wpk[L], Cout, Cin, LK[L] * LK[L], mode == 2, ntiles[L], nkt[L]);
        }

        const int M_total = Bn * Hout * Wout;
        const int mtiles  = (M_total + 15) / 16;
        const int ntg     = ntiles[L] / nblk;
        const int waves   = mtiles * ntg;
        const int blocks  = (waves + 7) / 8;
        const unsigned short* src = hbuf[cur];

        if (mode == 1) {
            conv_wmma_t<1, 7, 1, 3, 1><<<blocks, 256, 0, stream>>>(
                src, wpk[L], Bb[L], convOut, Bn, Cin, Hin, Win,
                Cout, Hout, Wout, mtiles, ntg, nkt[L]);
        } else if (mode == 0 && nblk == 2) {
            conv_wmma_t<0, 3, 2, 1, 2><<<blocks, 256, 0, stream>>>(
                src, wpk[L], Bb[L], convOut, Bn, Cin, Hin, Win,
                Cout, Hout, Wout, mtiles, ntg, nkt[L]);
        } else if (mode == 0) {
            conv_wmma_t<0, 3, 2, 1, 4><<<blocks, 256, 0, stream>>>(
                src, wpk[L], Bb[L], convOut, Bn, Cin, Hin, Win,
                Cout, Hout, Wout, mtiles, ntg, nkt[L]);
        } else if (nblk == 4) {
            conv_wmma_t<2, 3, 2, 1, 4><<<blocks, 256, 0, stream>>>(
                src, wpk[L], Bb[L], convOut, Bn, Cin, Hin, Win,
                Cout, Hout, Wout, mtiles, ntg, nkt[L]);
        } else if (nblk == 2) {
            conv_wmma_t<2, 3, 2, 1, 2><<<blocks, 256, 0, stream>>>(
                src, wpk[L], Bb[L], convOut, Bn, Cin, Hin, Win,
                Cout, Hout, Wout, mtiles, ntg, nkt[L]);
        } else {
            conv_wmma_t<2, 3, 2, 1, 1><<<blocks, 256, 0, stream>>>(
                src, wpk[L], Bb[L], convOut, Bn, Cin, Hin, Win,
                Cout, Hout, Wout, mtiles, ntg, nkt[L]);
        }

        const long tot = (long)Bn * Cout * Hout * Wout;
        if (L < 9) {
            inorm_stat_kernel<<<Bn * Cout, 256, 0, stream>>>(convOut, stats, Hout * Wout);
            inorm_relu_h16_kernel<<<(int)((tot + 255) / 256), 256, 0, stream>>>(
                convOut, stats, hbuf[cur ^ 1], Hout * Wout, tot);
            cur ^= 1;
        } else {
            tanh_kernel<<<(int)((tot + 255) / 256), 256, 0, stream>>>(convOut, tot);
        }
    }

    // ---- segment mean over instance labels
    const int HW = 512 * 512;
    seg_zero_kernel<<<1, 256, 0, stream>>>(seg);
    seg_acc_kernel<<<1024, 256, 0, stream>>>(convOut, inst, seg, Bn, HW);
    {
        long tot = (long)Bn * 3 * HW;
        seg_out_kernel<<<(int)((tot + 255) / 256), 256, 0, stream>>>(
            inst, seg, (float*)d_out, Bn, HW);
    }
}
